// MMD_65867618452140
// MI455X (gfx1250) — matrix operand order repaired and checked by reading, not compiled, not run on hardware
//
#include <hip/hip_runtime.h>

typedef __attribute__((ext_vector_type(16))) __bf16 v16bf;
typedef __attribute__((ext_vector_type(8)))  float  v8f;

#define NROWS 8192
#define DDIM  512
#define KC    32
#define NSTG  (DDIM / KC)   // 16 K-stages
#define GBLK  64            // 8192 / 128 macro-tiles per dim
// u16 elements per fragment-ordered copy of one matrix (hi+lo interleaved)
#define FRAGELEMS ((size_t)2 * NROWS * DDIM)
// u16 elements per (macroRow, stage) staging chunk: [hi/lo][tile0..7][512]
#define STG_ELEMS 8192
// u16 elements per macroRow: NSTG stages
#define MROW_ELEMS ((size_t)NSTG * STG_ELEMS)

// round-to-nearest-even float -> bf16 (raw u16); inputs are finite
__device__ __forceinline__ unsigned short f2bf(float x) {
    unsigned u = __float_as_uint(x);
    u += 0x7fffu + ((u >> 16) & 1u);
    return (unsigned short)(u >> 16);
}

// ---------------------------------------------------------------------------
// Prep: row norms (exact fp32) + one-time fp32 -> split-bf16 conversion into
// fragment-ordered global copies (A-operand order and B-operand order).
// Layout (u16 idx): (((macroRow*NSTG + stage)*2 + hl)*8 + tile)*512 + lane*16 + e
// ---------------------------------------------------------------------------
__global__ __launch_bounds__(128)
void mmd_prep(const float* __restrict__ X, const float* __restrict__ Y,
              unsigned short* __restrict__ XA, unsigned short* __restrict__ XB,
              unsigned short* __restrict__ YA, unsigned short* __restrict__ YB,
              float* __restrict__ xn, float* __restrict__ yn) {
    const int rb = blockIdx.x;
    const float* src;
    unsigned short *fa, *fb;
    float* dst;
    int r;
    if (rb < NROWS) { r = rb;         src = X + (size_t)r * DDIM; fa = XA; fb = XB; dst = xn + r; }
    else            { r = rb - NROWS; src = Y + (size_t)r * DDIM; fa = YA; fb = YB; dst = yn + r; }

    const int k = threadIdx.x << 2;                 // this thread's float4 (K offset)
    float4 f = ((const float4*)src)[threadIdx.x];

    // --- exact fp32 row norm ---
    float s = f.x * f.x + f.y * f.y + f.z * f.z + f.w * f.w;
#pragma unroll
    for (int off = 16; off > 0; off >>= 1) s += __shfl_xor(s, off, 32);
    __shared__ float sh[4];
    if ((threadIdx.x & 31) == 0) sh[threadIdx.x >> 5] = s;
    __syncthreads();
    if (threadIdx.x == 0) *dst = sh[0] + sh[1] + sh[2] + sh[3];

    // --- split bf16 conversion ---
    float xs[4] = {f.x, f.y, f.z, f.w};
    unsigned short h[4], l[4];
#pragma unroll
    for (int q = 0; q < 4; ++q) {
        h[q] = f2bf(xs[q]);
        float hf = __uint_as_float(((unsigned)h[q]) << 16);
        l[q] = f2bf(xs[q] - hf);
    }
    uint2 vh, vl;
    vh.x = (unsigned)h[0] | ((unsigned)h[1] << 16);
    vh.y = (unsigned)h[2] | ((unsigned)h[3] << 16);
    vl.x = (unsigned)l[0] | ((unsigned)l[1] << 16);
    vl.y = (unsigned)l[2] | ((unsigned)l[3] << 16);

    const int stage = k >> 5, kl = k & 31;
    const int mr = r >> 7, t = (r >> 4) & 7, m = r & 15;
    const size_t chunk = ((size_t)mr * NSTG + stage) * 2;   // in STG/2 units (hl slot)

    // A-operand fragment order: half = k[3], e = (k&7) + 8*(k>=16)
    {
        const int half = (kl >> 3) & 1;
        const int e0   = (kl & 7) | ((kl & 16) >> 1);
        const size_t fr = (half * 16 + m) * 16 + e0;
        *(uint2*)(fa + ((chunk + 0) * 8 + t) * 512 + fr) = vh;
        *(uint2*)(fa + ((chunk + 1) * 8 + t) * 512 + fr) = vl;
    }
    // B-operand fragment order: half = k[3], e = (k&7) + 8*(k>=16)
    {
        const int half = (kl >> 3) & 1;
        const int e0   = (kl & 7) | ((kl & 16) >> 1);
        const size_t fr = (half * 16 + m) * 16 + e0;
        *(uint2*)(fb + ((chunk + 0) * 8 + t) * 512 + fr) = vh;
        *(uint2*)(fb + ((chunk + 1) * 8 + t) * 512 + fr) = vl;
    }
}

// ---------------------------------------------------------------------------
// GEMM + fused exp: 128x128 macro-tile, 8 wave32 waves, double-buffered LDS
// fed by CDNA5 async global->LDS DMA (ASYNCcnt), 24 WMMA per wave per K-stage.
// ---------------------------------------------------------------------------
__global__ __launch_bounds__(256)
void mmd_gemm_exp(const unsigned short* __restrict__ Af,
                  const unsigned short* __restrict__ Bf,
                  const float* __restrict__ an, const float* __restrict__ bn,
                  float* __restrict__ partial) {
    __shared__ __align__(32) unsigned short sA[2][2][8][512];   // [buf][hi/lo][tile][512]
    __shared__ __align__(32) unsigned short sB[2][2][8][512];

    const int tid  = threadIdx.x;
    const int lane = tid & 31;
    const int wave = tid >> 5;
    const int wm   = wave >> 2;      // 0..1 : M-tiles wm*4 .. wm*4+3
    const int wn   = wave & 3;       // 0..3 : N-tiles wn*2 .. wn*2+1
    const int rowA0 = blockIdx.y * 128;
    const int rowB0 = blockIdx.x * 128;

    const unsigned short* Abase = Af + (size_t)blockIdx.y * MROW_ELEMS;
    const unsigned short* Bbase = Bf + (size_t)blockIdx.x * MROW_ELEMS;

    // async-copy one 16KB A chunk + 16KB B chunk (stage s) into LDS buffer `buf`
    auto stage_async = [&](int s, int buf) {
        const unsigned short* ga = Abase + (size_t)s * STG_ELEMS;
        const unsigned short* gb = Bbase + (size_t)s * STG_ELEMS;
        const unsigned la = (unsigned)(unsigned long long)&sA[buf][0][0][0];
        const unsigned lb = (unsigned)(unsigned long long)&sB[buf][0][0][0];
#pragma unroll
        for (int c = 0; c < 4; ++c) {
            const int o = (tid + 256 * c) * 8;           // u16 units, 16B chunks
            asm volatile("global_load_async_to_lds_b128 %0, %1, off"
                         :: "v"(la + (unsigned)(o * 2)), "v"(ga + o) : "memory");
            asm volatile("global_load_async_to_lds_b128 %0, %1, off"
                         :: "v"(lb + (unsigned)(o * 2)), "v"(gb + o) : "memory");
        }
    };

    v8f acc[4][2];
#pragma unroll
    for (int a = 0; a < 4; ++a)
#pragma unroll
        for (int b = 0; b < 2; ++b) { v8f z = {}; acc[a][b] = z; }

    stage_async(0, 0);
    asm volatile("s_wait_asynccnt 0x0" ::: "memory");
    __syncthreads();

    for (int s = 0; s < NSTG; ++s) {
        const int cur = s & 1, nxt = cur ^ 1;
        if (s + 1 < NSTG) stage_async(s + 1, nxt);      // overlap DMA with WMMA

        // load all fragments for this stage, then issue all WMMAs
        v16bf ahi[4], alo[4], bhi[2], blo[2];
#pragma unroll
        for (int b = 0; b < 2; ++b) {
            const int t = wn * 2 + b;
            bhi[b] = *(const v16bf*)&sB[cur][0][t][lane * 16];
            blo[b] = *(const v16bf*)&sB[cur][1][t][lane * 16];
        }
#pragma unroll
        for (int a = 0; a < 4; ++a) {
            const int t = wm * 4 + a;
            ahi[a] = *(const v16bf*)&sA[cur][0][t][lane * 16];
            alo[a] = *(const v16bf*)&sA[cur][1][t][lane * 16];
        }
        // dot += Ahi*Bhi + Ahi*Blo + Alo*Bhi  (fp32-accurate split-bf16)
#pragma unroll
        for (int a = 0; a < 4; ++a)
#pragma unroll
            for (int b = 0; b < 2; ++b) {
                acc[a][b] = __builtin_amdgcn_wmma_f32_16x16x32_bf16(
                    false, ahi[a], false, bhi[b], (short)0, acc[a][b], false, false);
                acc[a][b] = __builtin_amdgcn_wmma_f32_16x16x32_bf16(
                    false, ahi[a], false, blo[b], (short)0, acc[a][b], false, false);
                acc[a][b] = __builtin_amdgcn_wmma_f32_16x16x32_bf16(
                    false, alo[a], false, bhi[b], (short)0, acc[a][b], false, false);
            }

        if (s + 1 < NSTG) asm volatile("s_wait_asynccnt 0x0" ::: "memory");
        __syncthreads();
    }

    // epilogue: C/D layout -> m = v + 8*(lane>>4), n = lane&15
    const float invBW = 1.0f / 512.0f;
    const int half = lane >> 4, nidx = lane & 15;
    float lsum = 0.0f;
#pragma unroll
    for (int a = 0; a < 4; ++a) {
        const int ibase = rowA0 + (wm * 4 + a) * 16 + 8 * half;
#pragma unroll
        for (int b = 0; b < 2; ++b) {
            const int j = rowB0 + (wn * 2 + b) * 16 + nidx;
            const float bnj = bn[j];
#pragma unroll
            for (int v = 0; v < 8; ++v) {
                float d2 = an[ibase + v] + bnj - 2.0f * acc[a][b][v];
                lsum += __expf(-d2 * invBW);
            }
        }
    }
#pragma unroll
    for (int off = 16; off > 0; off >>= 1) lsum += __shfl_xor(lsum, off, 32);

    float* red = (float*)&sA[0][0][0][0];   // LDS reuse (last loop iter ends in barrier)
    if (lane == 0) red[wave] = lsum;
    __syncthreads();
    if (tid == 0) {
        float t = 0.0f;
#pragma unroll
        for (int w = 0; w < 8; ++w) t += red[w];
        partial[blockIdx.y * gridDim.x + blockIdx.x] = t;
    }
}

__global__ __launch_bounds__(256)
void mmd_finalize(const float* __restrict__ partial, float* __restrict__ out) {
    __shared__ double sh[256];
    const int tid = threadIdx.x;
    double S[3];
    for (int p = 0; p < 3; ++p) {
        double s = 0.0;
        for (int i = tid; i < GBLK * GBLK; i += 256) s += (double)partial[p * GBLK * GBLK + i];
        sh[tid] = s;
        __syncthreads();
        for (int off = 128; off > 0; off >>= 1) {
            if (tid < off) sh[tid] += sh[tid + off];
            __syncthreads();
        }
        S[p] = sh[0];
        __syncthreads();
    }
    if (tid == 0) {
        const double n = (double)NROWS, m = (double)NROWS;
        double xx = (S[0] - n) / (n * (n - 1.0));
        double xy = S[1] / (n * m);
        double yy = (S[2] - m) / (m * (m - 1.0));
        out[0] = (float)(xx - 2.0 * xy + yy);
    }
}

extern "C" void kernel_launch(void* const* d_in, const int* in_sizes, int n_in,
                              void* d_out, int out_size, void* d_ws, size_t ws_size,
                              hipStream_t stream) {
    (void)in_sizes; (void)n_in; (void)out_size; (void)ws_size;
    const float* X = (const float*)d_in[0];
    const float* Y = (const float*)d_in[1];

    unsigned short* XA = (unsigned short*)d_ws;   // 16MB each (hi+lo, frag-ordered)
    unsigned short* XB = XA + FRAGELEMS;
    unsigned short* YA = XB + FRAGELEMS;
    unsigned short* YB = YA + FRAGELEMS;
    float* xn   = (float*)(YB + FRAGELEMS);       // 8192 floats
    float* yn   = xn + NROWS;                     // 8192 floats
    float* part = yn + NROWS;                     // 3 * 4096 floats

    mmd_prep<<<2 * NROWS, 128, 0, stream>>>(X, Y, XA, XB, YA, YB, xn, yn);
    dim3 g(GBLK, GBLK);
    mmd_gemm_exp<<<g, 256, 0, stream>>>(XA, XB, xn, xn, part);
    mmd_gemm_exp<<<g, 256, 0, stream>>>(XA, YB, xn, yn, part + GBLK * GBLK);
    mmd_gemm_exp<<<g, 256, 0, stream>>>(YA, YB, yn, yn, part + 2 * GBLK * GBLK);
    mmd_finalize<<<1, 256, 0, stream>>>(part, (float*)d_out);
}
